// MultiHeadAttention_10462540333415
// MI455X (gfx1250) — matrix operand order repaired and checked by reading, not compiled, not run on hardware
//
#include <hip/hip_runtime.h>

#define D_MODEL 1024
#define N_HEADS 16
#define D_HEAD  64
#define BATCH   4
#define SEQ     2048
#define M_ROWS  (BATCH * SEQ)   // 8192

typedef __attribute__((ext_vector_type(16))) __bf16 v16bf;
typedef __attribute__((ext_vector_type(8)))  __bf16 bf16x8;
typedef __attribute__((ext_vector_type(8)))  float  v8f;

#define NEG_INF (-__builtin_inff())

// store modes for the GEMM epilogue
#define ST_BF16_ROWMAJOR 0   // bf16 [M,N]
#define ST_F32_ROWMAJOR  1   // f32  [M,N]
#define ST_BF16_HEADMAJ  2   // bf16 [B,H,S,Dh]   (Q, K)
#define ST_BF16_HEADTR   3   // bf16 [B,H,Dh,S]   (V)

// ---------------------------------------------------------------------------
// fp32 -> bf16 conversion (grid-stride)
// ---------------------------------------------------------------------------
__global__ void cvt_f32_bf16(const float* __restrict__ src,
                             __bf16* __restrict__ dst, int n) {
  for (int i = blockIdx.x * blockDim.x + threadIdx.x; i < n;
       i += gridDim.x * blockDim.x) {
    dst[i] = (__bf16)src[i];
  }
}

// W [K x N] fp32 row-major  ->  Wt [N x K] bf16 (K contiguous per column)
__global__ void cvt_w_transpose(const float* __restrict__ W,
                                __bf16* __restrict__ Wt) {
  int idx = blockIdx.x * blockDim.x + threadIdx.x;   // over 1024*1024
  if (idx >= D_MODEL * D_MODEL) return;
  int k = idx >> 10;
  int n = idx & (D_MODEL - 1);
  Wt[(size_t)n * D_MODEL + k] = (__bf16)W[idx];
}

// ---------------------------------------------------------------------------
// Fragment helpers (wave32; this code's contiguous K order, same for A and B)
//   A 16x32: lane row = lane&15; elems 0..15 -> K = 16*(lane>>4)+e
//   B 32x16: lane col = lane&15; elems 0..15 -> K = 16*(lane>>4)+e
// ---------------------------------------------------------------------------
__device__ __forceinline__ v16bf load_a_frag(const __bf16* row_ptr, int k0, int hi) {
  bf16x8 lo = *(const bf16x8*)(row_ptr + k0 + hi * 16);
  bf16x8 hv = *(const bf16x8*)(row_ptr + k0 + hi * 16 + 8);
  v16bf a;
#pragma unroll
  for (int i = 0; i < 8; ++i) { a[i] = lo[i]; a[i + 8] = hv[i]; }
  return a;
}

__device__ __forceinline__ v16bf load_b_frag(const __bf16* kptr) {
  // kptr already points at this lane's column, at K-offset 16*(lane>>4)
  bf16x8 lo = *(const bf16x8*)(kptr);
  bf16x8 hv = *(const bf16x8*)(kptr + 8);
  v16bf b;
#pragma unroll
  for (int i = 0; i < 8; ++i) { b[i] = lo[i]; b[i + 8] = hv[i]; }
  return b;
}

// ---------------------------------------------------------------------------
// Register-blocked bf16 WMMA GEMM: C[M,N] = A[M,K] * Bt[N,K]^T + bias
// One wave -> 64x32 output (4 M-tiles x 2 N-tiles): 8 WMMAs per 12 b128
// loads per K-step.  Block = 128 threads = 4 waves stacked along M;
// grid = (N/32, M/256).
// ---------------------------------------------------------------------------
__global__ void gemm_bf16(const __bf16* __restrict__ A,
                          const __bf16* __restrict__ Bt,
                          const float* __restrict__ bias,
                          void* __restrict__ out, int mode,
                          int M, int N, int Kd) {
  int lane = threadIdx.x & 31;
  int wave = threadIdx.x >> 5;
  int hi = lane >> 4;
  int lr = lane & 15;
  int nbase = blockIdx.x * 32;
  int mbase = (blockIdx.y * 4 + wave) * 64;

  const __bf16* arow[4];
  const __bf16* brow[2];
#pragma unroll
  for (int i = 0; i < 4; ++i) arow[i] = A + (size_t)(mbase + i * 16 + lr) * Kd;
#pragma unroll
  for (int j = 0; j < 2; ++j) brow[j] = Bt + (size_t)(nbase + j * 16 + lr) * Kd;

  v8f acc[4][2];
#pragma unroll
  for (int i = 0; i < 4; ++i)
#pragma unroll
    for (int j = 0; j < 2; ++j) acc[i][j] = (v8f){};

#pragma unroll 2
  for (int k0 = 0; k0 < Kd; k0 += 32) {
    v16bf af[4], bfv[2];
#pragma unroll
    for (int i = 0; i < 4; ++i) {
      __builtin_prefetch(arow[i] + k0 + 256, 0, 3);
      af[i] = load_a_frag(arow[i], k0, hi);
    }
#pragma unroll
    for (int j = 0; j < 2; ++j) bfv[j] = load_b_frag(brow[j] + k0 + hi * 16);
#pragma unroll
    for (int i = 0; i < 4; ++i)
#pragma unroll
      for (int j = 0; j < 2; ++j)
        acc[i][j] = __builtin_amdgcn_wmma_f32_16x16x32_bf16(
            false, af[i], false, bfv[j], (short)0, acc[i][j], false, false);
  }

  float bn[2];
#pragma unroll
  for (int j = 0; j < 2; ++j)
    bn[j] = bias ? bias[nbase + j * 16 + lr] : 0.0f;

  __bf16* ob = (__bf16*)out;
  float* of = (float*)out;
#pragma unroll
  for (int i = 0; i < 4; ++i) {
#pragma unroll
    for (int j = 0; j < 2; ++j) {
#pragma unroll
      for (int v = 0; v < 8; ++v) {
        int m = mbase + i * 16 + hi * 8 + v;     // global row (b*S + s)
        int n = nbase + j * 16 + lr;             // global col (h*Dh + d)
        float val = acc[i][j][v] + bn[j];
        if (mode == ST_F32_ROWMAJOR) {
          of[(size_t)m * N + n] = val;
        } else if (mode == ST_BF16_ROWMAJOR) {
          ob[(size_t)m * N + n] = (__bf16)val;
        } else {
          int b = m >> 11;                // / SEQ
          int s = m & (SEQ - 1);
          int h = n >> 6;                 // / D_HEAD
          int d = n & (D_HEAD - 1);
          size_t idx;
          if (mode == ST_BF16_HEADMAJ)    // [B,H,S,Dh]
            idx = ((size_t)(b * N_HEADS + h) * SEQ + s) * D_HEAD + d;
          else                            // [B,H,Dh,S]
            idx = ((size_t)(b * N_HEADS + h) * D_HEAD + d) * SEQ + s;
          ob[idx] = (__bf16)val;
        }
      }
    }
  }
}

// ---------------------------------------------------------------------------
// Flash attention: one wave per (b, h, 16-query tile).  32 keys / iteration.
// Qh,Kh: [B,H,S,Dh] bf16;  Vt: [B,H,Dh,S] bf16;  mask: [B,S] bytes;
// O: [B,S,H*Dh] bf16.
// ---------------------------------------------------------------------------
__global__ void attn_kernel(const __bf16* __restrict__ Qh,
                            const __bf16* __restrict__ Kh,
                            const __bf16* __restrict__ Vt,
                            const unsigned char* __restrict__ msk,
                            __bf16* __restrict__ O) {
  __shared__ __bf16 ldsP[4][16 * 32];   // per-wave P tile (16 q x 32 keys)

  int lane = threadIdx.x & 31;
  int wave = threadIdx.x >> 5;
  int hi = lane >> 4;
  int lr = lane & 15;

  int task = blockIdx.x * 4 + wave;        // 0 .. B*H*(S/16)-1
  int qt = task & 127;                     // S/16 = 128
  int h = (task >> 7) & (N_HEADS - 1);
  int b = task >> 11;

  const size_t headQ = (size_t)(b * N_HEADS + h) * SEQ;    // row base (seq-major)
  const size_t headV = (size_t)(b * N_HEADS + h) * D_HEAD; // row base (d-major)

  // Q fragments for both 32-wide d_head chunks (loaded once)
  const __bf16* qrow = Qh + (headQ + qt * 16 + lr) * D_HEAD;
  v16bf qa0 = load_a_frag(qrow, 0, hi);
  v16bf qa1 = load_a_frag(qrow, 32, hi);

  v8f o[4];
#pragma unroll
  for (int t = 0; t < 4; ++t) o[t] = (v8f){};
  float m[8], l[8];
#pragma unroll
  for (int v = 0; v < 8; ++v) { m[v] = NEG_INF; l[v] = 0.0f; }

  const unsigned char* mrow = msk + (size_t)b * SEQ;
  __bf16* pw = ldsP[wave];
  const float scale = 0.125f;   // 1/sqrt(64)

  for (int kb = 0; kb < SEQ; kb += 32) {
    // ---- raw scores: two 16-key tiles, Kdim = 64 (2 chunks of 32) ----
    v8f s0 = {}, s1 = {};
    {
      const __bf16* kb0 = Kh + (headQ + kb + lr) * D_HEAD;
      const __bf16* kb1 = Kh + (headQ + kb + 16 + lr) * D_HEAD;
      v16bf f;
      f = load_b_frag(kb0 + 0 + hi * 16);
      s0 = __builtin_amdgcn_wmma_f32_16x16x32_bf16(false, qa0, false, f, (short)0, s0, false, false);
      f = load_b_frag(kb1 + 0 + hi * 16);
      s1 = __builtin_amdgcn_wmma_f32_16x16x32_bf16(false, qa0, false, f, (short)0, s1, false, false);
      f = load_b_frag(kb0 + 32 + hi * 16);
      s0 = __builtin_amdgcn_wmma_f32_16x16x32_bf16(false, qa1, false, f, (short)0, s0, false, false);
      f = load_b_frag(kb1 + 32 + hi * 16);
      s1 = __builtin_amdgcn_wmma_f32_16x16x32_bf16(false, qa1, false, f, (short)0, s1, false, false);
    }

    // ---- mask (-inf BEFORE scale, matching reference), online softmax ----
    bool ok0 = mrow[kb + lr] != 0;
    bool ok1 = mrow[kb + 16 + lr] != 0;
    float x0[8], x1[8], mn[8], al[8], p0[8], p1[8];
#pragma unroll
    for (int v = 0; v < 8; ++v) {
      x0[v] = ok0 ? s0[v] * scale : NEG_INF;
      x1[v] = ok1 ? s1[v] * scale : NEG_INF;
    }
#pragma unroll
    for (int v = 0; v < 8; ++v) {
      float rv = fmaxf(x0[v], x1[v]);
#pragma unroll
      for (int off = 1; off < 16; off <<= 1)
        rv = fmaxf(rv, __shfl_xor(rv, off, 32));
      mn[v] = fmaxf(m[v], rv);
      al[v] = (mn[v] == NEG_INF) ? 1.0f : __expf(m[v] - mn[v]);
    }
#pragma unroll
    for (int v = 0; v < 8; ++v) {
      p0[v] = (x0[v] == NEG_INF) ? 0.0f : __expf(x0[v] - mn[v]);
      p1[v] = (x1[v] == NEG_INF) ? 0.0f : __expf(x1[v] - mn[v]);
      float sv = p0[v] + p1[v];
#pragma unroll
      for (int off = 1; off < 16; off <<= 1)
        sv += __shfl_xor(sv, off, 32);
      l[v] = l[v] * al[v] + sv;
      m[v] = mn[v];
    }
    // rescale running output
#pragma unroll
    for (int t = 0; t < 4; ++t)
#pragma unroll
      for (int v = 0; v < 8; ++v) o[t][v] *= al[v];

    // ---- stage P (C/D layout -> LDS row-major) and reload as A fragment ----
#pragma unroll
    for (int v = 0; v < 8; ++v) {
      int row = v + hi * 8;
      pw[row * 32 + lr] = (__bf16)p0[v];
      pw[row * 32 + 16 + lr] = (__bf16)p1[v];
    }
    v16bf pf = load_a_frag(pw + lr * 32, 0, hi);

    // ---- O += P @ V   (4 d_head tiles of 16) ----
#pragma unroll
    for (int t = 0; t < 4; ++t) {
      const __bf16* vb = Vt + (headV + t * 16 + lr) * SEQ + kb + hi * 16;
      v16bf f = load_b_frag(vb);
      o[t] = __builtin_amdgcn_wmma_f32_16x16x32_bf16(false, pf, false, f,
                                                     (short)0, o[t], false, false);
    }
  }

  // ---- epilogue: normalize and store bf16 in [B,S,H*Dh] ----
#pragma unroll
  for (int t = 0; t < 4; ++t) {
#pragma unroll
    for (int v = 0; v < 8; ++v) {
      int q = qt * 16 + v + hi * 8;
      float val = o[t][v] / l[v];
      O[(size_t)(b * SEQ + q) * D_MODEL + h * D_HEAD + t * 16 + lr] = (__bf16)val;
    }
  }
}

// ---------------------------------------------------------------------------
extern "C" void kernel_launch(void* const* d_in, const int* in_sizes, int n_in,
                              void* d_out, int out_size, void* d_ws, size_t ws_size,
                              hipStream_t stream) {
  (void)in_sizes; (void)n_in; (void)out_size; (void)ws_size;

  const float* x  = (const float*)d_in[0];
  const unsigned char* mask = (const unsigned char*)d_in[1];
  const float* Wq = (const float*)d_in[2]; const float* bq = (const float*)d_in[3];
  const float* Wk = (const float*)d_in[4]; const float* bk = (const float*)d_in[5];
  const float* Wv = (const float*)d_in[6]; const float* bv = (const float*)d_in[7];
  const float* Wo = (const float*)d_in[8]; const float* bo = (const float*)d_in[9];
  float* out = (float*)d_out;

  // workspace carve-up (bf16 buffers), ~88 MB total
  char* w = (char*)d_ws;
  const size_t XB = (size_t)M_ROWS * D_MODEL * 2;          // 16 MB
  const size_t WB = (size_t)D_MODEL * D_MODEL * 2;         // 2 MB
  __bf16* Xb   = (__bf16*)(w);
  __bf16* Wqt  = (__bf16*)(w + XB);
  __bf16* Wkt  = (__bf16*)(w + XB + WB);
  __bf16* Wvt  = (__bf16*)(w + XB + 2 * WB);
  __bf16* Wot  = (__bf16*)(w + XB + 3 * WB);
  __bf16* Qh   = (__bf16*)(w + XB + 4 * WB);               // [B,H,S,Dh]
  __bf16* Kh   = (__bf16*)(w + 2 * XB + 4 * WB);           // [B,H,S,Dh]
  __bf16* Vt   = (__bf16*)(w + 3 * XB + 4 * WB);           // [B,H,Dh,S]
  __bf16* Attn = (__bf16*)(w + 4 * XB + 4 * WB);           // [B,S,H*Dh]

  const int nX = M_ROWS * D_MODEL;
  cvt_f32_bf16<<<2048, 256, 0, stream>>>(x, Xb, nX);
  cvt_w_transpose<<<(D_MODEL * D_MODEL) / 256, 256, 0, stream>>>(Wq, Wqt);
  cvt_w_transpose<<<(D_MODEL * D_MODEL) / 256, 256, 0, stream>>>(Wk, Wkt);
  cvt_w_transpose<<<(D_MODEL * D_MODEL) / 256, 256, 0, stream>>>(Wv, Wvt);
  cvt_w_transpose<<<(D_MODEL * D_MODEL) / 256, 256, 0, stream>>>(Wo, Wot);

  dim3 ggrid(D_MODEL / 32, M_ROWS / 256);   // (32, 32), 128 threads (4 waves)
  gemm_bf16<<<ggrid, 128, 0, stream>>>(Xb, Wqt, bq, Qh, ST_BF16_HEADMAJ,
                                       M_ROWS, D_MODEL, D_MODEL);
  gemm_bf16<<<ggrid, 128, 0, stream>>>(Xb, Wkt, bk, Kh, ST_BF16_HEADMAJ,
                                       M_ROWS, D_MODEL, D_MODEL);
  gemm_bf16<<<ggrid, 128, 0, stream>>>(Xb, Wvt, bv, Vt, ST_BF16_HEADTR,
                                       M_ROWS, D_MODEL, D_MODEL);

  // B*H*(S/16) tasks / 4 waves per block = 2048 blocks
  attn_kernel<<<2048, 128, 0, stream>>>(Qh, Kh, Vt, mask, Attn);

  gemm_bf16<<<ggrid, 128, 0, stream>>>(Attn, Wot, bo, out, ST_F32_ROWMAJOR,
                                       M_ROWS, D_MODEL, D_MODEL);
}
